// BEVEncoder_71373766525402
// MI455X (gfx1250) — matrix operand order repaired and checked by reading, not compiled, not run on hardware
//
#include <hip/hip_runtime.h>
#include <math.h>

// ---------------------------------------------------------------------------
// BEVFormer-style encoder for gfx1250 (MI455X).
// GEMMs: v_wmma_f32_16x16x32_f16, f16 operands staged in LDS via
// global_load_async_to_lds_b128 with a 2-stage double-buffered pipeline
// (partial s_wait_asynccnt), f32 accumulate.
// ---------------------------------------------------------------------------

#define NQ        (128 * 128)
#define BEVH      128
#define BEVW      128
#define CDIM      256
#define HEADS     8
#define HD        32
#define NPTS      4
#define NLAYERS   6
#define NCAM      6
#define HF        58
#define WF        100
#define FEAT_ROWS (NCAM * HF * WF)        // 34800
#define FEAT_PAD  34816                   // rounded up to 64
#define HP        (HEADS * NPTS)          // 32
#define HP2       (HP * 2)                // 64

typedef _Float16 v16h __attribute__((ext_vector_type(16)));
typedef _Float16 v8h  __attribute__((ext_vector_type(8)));
typedef float    v8f  __attribute__((ext_vector_type(8)));
typedef int      v4i  __attribute__((vector_size(16)));   // matches async builtin

union Frag16 { v16h v; v8h h[2]; };

#define AS1 __attribute__((address_space(1)))
#define AS3 __attribute__((address_space(3)))

#if defined(__AMDGCN__) && __has_builtin(__builtin_amdgcn_global_load_async_to_lds_b128)
#define USE_ASYNC_LDS 1
#else
#define USE_ASYNC_LDS 0
#endif

#if USE_ASYNC_LDS
__device__ __forceinline__ void async_cp16(const _Float16* src, _Float16* ldsDst)
{
    __builtin_amdgcn_global_load_async_to_lds_b128(
        (AS1 v4i*)src, (AS3 v4i*)ldsDst, 0, 0);
}
__device__ __forceinline__ void wait_async0()
{
#if __has_builtin(__builtin_amdgcn_s_wait_asynccnt)
    __builtin_amdgcn_s_wait_asynccnt(0);
#else
    asm volatile("s_wait_asynccnt 0" ::: "memory");
#endif
}
__device__ __forceinline__ void wait_async4()
{
#if __has_builtin(__builtin_amdgcn_s_wait_asynccnt)
    __builtin_amdgcn_s_wait_asynccnt(4);
#else
    asm volatile("s_wait_asynccnt 4" ::: "memory");
#endif
}
#endif

// ---------------------------------------------------------------------------
// GEMM: out[M,N] = act(A[M,K] @ Wt[N,K]^T + bias[N]) ; A,Wt f16 ; out f32 or f16
// Block tile 64x64, 4 waves, wave tile 32x32 (2x2 WMMA frags). M % 64 == 0.
// NGUARD instantiation handles N not a multiple of 64 (attention logits, N=32).
// ---------------------------------------------------------------------------
#define TM 64
#define TN 64
#define TK 32
#define LDSP (TK + 8)   // 40 halves = 80B row stride, 16B aligned

template<bool NGUARD>
__global__ __launch_bounds__(128)
void gemm_wmma_f16(const _Float16* __restrict__ A, const _Float16* __restrict__ Wt,
                   const float* __restrict__ bias, float* __restrict__ Cf,
                   _Float16* __restrict__ Ch, int M, int N, int K, int relu)
{
    __shared__ __align__(16) _Float16 sA[2][TM][LDSP];
    __shared__ __align__(16) _Float16 sB[2][TN][LDSP];   // B stored as [n][k]

    const int tid  = threadIdx.x;
    const int wave = tid >> 5;
    const int lane = tid & 31;
    const int m0   = blockIdx.y * TM;
    const int n0   = blockIdx.x * TN;
    const int wm   = (wave >> 1) * 32;
    const int wn   = (wave & 1) * 32;
    const int lh   = lane >> 4;
    const int lm   = lane & 15;

    // staging: 16B (8-half) chunks; 4 chunks per 32-wide row; 256 chunks/tile;
    // each of the 128 threads owns chunk tid and chunk tid+128.
    const int r0 = tid >> 2;
    const int o0 = (tid & 3) << 3;
    const int r1 = (tid + 128) >> 2;
    const int o1 = ((tid + 128) & 3) << 3;
    const _Float16* gA0 = A  + (size_t)(m0 + r0) * K + o0;
    const _Float16* gA1 = A  + (size_t)(m0 + r1) * K + o1;
    const _Float16* gB0 = Wt + (size_t)(n0 + r0) * K + o0;
    const _Float16* gB1 = Wt + (size_t)(n0 + r1) * K + o1;
    const bool bv0 = !NGUARD || (n0 + r0 < N);
    const bool bv1 = !NGUARD || (n0 + r1 < N);

    v8f acc[2][2] = {};

    // A fragment (16x32): half 0 -> K0..15 ; half 1 -> K16..31
    // B fragment (32x16): half 0 -> K0..15 ; half 1 -> K16..31
    auto compute_tile = [&](int buf) {
        Frag16 af[2], bf[2];
        #pragma unroll
        for (int i = 0; i < 2; i++) {
            int row = wm + i * 16 + lm;
            af[i].h[0] = *(const v8h*)&sA[buf][row][lh * 16];
            af[i].h[1] = *(const v8h*)&sA[buf][row][lh * 16 + 8];
        }
        #pragma unroll
        for (int j = 0; j < 2; j++) {
            int col = wn + j * 16 + lm;
            bf[j].h[0] = *(const v8h*)&sB[buf][col][lh * 16];
            bf[j].h[1] = *(const v8h*)&sB[buf][col][lh * 16 + 8];
        }
        #pragma unroll
        for (int i = 0; i < 2; i++)
            #pragma unroll
            for (int j = 0; j < 2; j++)
                acc[i][j] = __builtin_amdgcn_wmma_f32_16x16x32_f16(
                    false, af[i].v, false, bf[j].v,
                    (short)0, acc[i][j], false, false);
    };

#if USE_ASYNC_LDS
    if (!NGUARD) {
        // 2-stage software pipeline on ASYNCcnt: tile k+1 streams into the
        // other LDS buffer while tile k is computed; partial wait (<=4)
        // retires only the current tile's 4 copies.
        auto issue_tile = [&](int k0, int buf) {
            async_cp16(gA0 + k0, &sA[buf][r0][o0]);
            async_cp16(gA1 + k0, &sA[buf][r1][o1]);
            async_cp16(gB0 + k0, &sB[buf][r0][o0]);
            async_cp16(gB1 + k0, &sB[buf][r1][o1]);
        };
        const int KT = K / TK;
        issue_tile(0, 0);
        for (int kt = 0; kt < KT; kt++) {
            const int buf = kt & 1;
            if (kt + 1 < KT) {
                issue_tile((kt + 1) * TK, (kt + 1) & 1);
                wait_async4();          // current tile's copies retired
            } else {
                wait_async0();
            }
            __syncthreads();            // all waves' current tile visible
            compute_tile(buf);
            __syncthreads();            // frees buf for tile kt+2
        }
    } else
#endif
    {
        for (int k0 = 0; k0 < K; k0 += TK) {
            v8h a0 = *(const v8h*)(gA0 + k0);
            v8h a1 = *(const v8h*)(gA1 + k0);
            v8h b0 = {}, b1 = {};
            if (bv0) b0 = *(const v8h*)(gB0 + k0);
            if (bv1) b1 = *(const v8h*)(gB1 + k0);
            __syncthreads();
            *(v8h*)&sA[0][r0][o0] = a0;
            *(v8h*)&sA[0][r1][o1] = a1;
            *(v8h*)&sB[0][r0][o0] = b0;
            *(v8h*)&sB[0][r1][o1] = b1;
            __syncthreads();
            compute_tile(0);
        }
    }

    // epilogue: lane -> col lm, acc elem e -> row e + 8*lh
    #pragma unroll
    for (int i = 0; i < 2; i++) {
        #pragma unroll
        for (int j = 0; j < 2; j++) {
            int gn = n0 + wn + j * 16 + lm;
            if (NGUARD && gn >= N) continue;
            float bv = bias[gn];
            #pragma unroll
            for (int e = 0; e < 8; e++) {
                int gm = m0 + wm + i * 16 + lh * 8 + e;
                float v = acc[i][j][e] + bv;
                if (relu) v = fmaxf(v, 0.f);
                if (Ch) Ch[(size_t)gm * N + gn] = (_Float16)v;
                else    Cf[(size_t)gm * N + gn] = v;
            }
        }
    }
}

static inline void run_gemm(const _Float16* A, const _Float16* Wt, const float* bias,
                            float* Cf, _Float16* Ch, int M, int N, int K, int relu,
                            hipStream_t s)
{
    dim3 grid((N + TN - 1) / TN, M / TM);
    if ((N % TN) == 0)
        gemm_wmma_f16<false><<<grid, 128, 0, s>>>(A, Wt, bias, Cf, Ch, M, N, K, relu);
    else
        gemm_wmma_f16<true><<<grid, 128, 0, s>>>(A, Wt, bias, Cf, Ch, M, N, K, relu);
}

// ---------------------------------------------------------------------------
// One-time cast / layout kernels
// ---------------------------------------------------------------------------
__global__ __launch_bounds__(256)
void cast_f32_to_f16(const float* __restrict__ in, _Float16* __restrict__ out, int n)
{
    int i = blockIdx.x * 256 + threadIdx.x;
    if (i < n) out[i] = (_Float16)in[i];
}

// W[K,N] f32 -> Wt[N,K] f16
__global__ __launch_bounds__(256)
void cast_transpose_w(const float* __restrict__ W, _Float16* __restrict__ Wt,
                      int K, int N)
{
    int i = blockIdx.x * 256 + threadIdx.x;
    if (i >= N * K) return;
    int n = i / K, k = i - n * K;
    Wt[i] = (_Float16)W[(size_t)k * N + n];
}

// image_feats (Ncam,C,Hf,Wf) f32 -> feats_h (FEAT_PAD, C) f16, zero-padded rows
__global__ __launch_bounds__(256)
void feats_to_flat_h(const float* __restrict__ in, _Float16* __restrict__ out)
{
    int idx = blockIdx.x * 256 + threadIdx.x;
    if (idx >= FEAT_PAD * CDIM) return;
    int ch = idx & (CDIM - 1);
    int m  = idx >> 8;
    float v = 0.f;
    if (m < FEAT_ROWS) {
        int x = m % WF;
        int y = (m / WF) % HF;
        int c = m / (WF * HF);
        v = in[(((size_t)c * CDIM + ch) * HF + y) * WF + x];
    }
    out[idx] = (_Float16)v;
}

// ---------------------------------------------------------------------------
// Invert 4x4 extrinsics (Gauss-Jordan with partial pivoting), 6 cameras.
// ---------------------------------------------------------------------------
__global__ void invert_ext(const float* __restrict__ extr, float* __restrict__ inv)
{
    int c = threadIdx.x;
    if (c >= NCAM) return;
    float a[4][8];
    for (int i = 0; i < 4; i++)
        for (int j = 0; j < 4; j++) {
            a[i][j]     = extr[c * 16 + i * 4 + j];
            a[i][4 + j] = (i == j) ? 1.f : 0.f;
        }
    for (int col = 0; col < 4; col++) {
        int piv = col;
        for (int r = col + 1; r < 4; r++)
            if (fabsf(a[r][col]) > fabsf(a[piv][col])) piv = r;
        if (piv != col)
            for (int j = 0; j < 8; j++) {
                float t = a[col][j]; a[col][j] = a[piv][j]; a[piv][j] = t;
            }
        float d = 1.f / a[col][col];
        for (int j = 0; j < 8; j++) a[col][j] *= d;
        for (int r = 0; r < 4; r++) {
            if (r == col) continue;
            float f = a[r][col];
            for (int j = 0; j < 8; j++) a[r][j] -= f * a[col][j];
        }
    }
    for (int i = 0; i < 4; i++)
        for (int j = 0; j < 4; j++)
            inv[c * 16 + i * 4 + j] = a[i][4 + j];
}

// ---------------------------------------------------------------------------
// Project 3D reference points to cameras -> ref_cam (c,n,p,2), valid (c,n,p)
// ---------------------------------------------------------------------------
__global__ __launch_bounds__(256)
void project_refs(const float* __restrict__ ego2cam, const float* __restrict__ intr,
                  float* __restrict__ ref_cam, float* __restrict__ valid)
{
    int idx = blockIdx.x * 256 + threadIdx.x;
    if (idx >= NCAM * NQ * NPTS) return;
    int p = idx & (NPTS - 1);
    int n = (idx >> 2) & (NQ - 1);
    int c = idx >> 16;               // NQ*NPTS = 65536
    int qx = n & (BEVW - 1), qy = n >> 7;
    float gx = (qx + 0.5f) / (float)BEVW;
    float gy = (qy + 0.5f) / (float)BEVH;
    float xm = -51.2f + gx * 102.4f;
    float ym = -51.2f + gy * 102.4f;
    float zm = -5.0f + (p + 0.5f) * (1.f / NPTS) * 8.0f;
    const float* E = ego2cam + c * 16;
    float pc[3];
    #pragma unroll
    for (int i = 0; i < 3; i++)
        pc[i] = E[i*4+0]*xm + E[i*4+1]*ym + E[i*4+2]*zm + E[i*4+3];
    const float* I = intr + c * 9;
    float u = I[0]*pc[0] + I[1]*pc[1] + I[2]*pc[2];
    float v = I[3]*pc[0] + I[4]*pc[1] + I[5]*pc[2];
    float z = I[6]*pc[0] + I[7]*pc[1] + I[8]*pc[2];
    float zc = fmaxf(z, 1e-5f);
    float un = u / (zc * (float)WF);
    float vn = v / (zc * (float)HF);
    float ok = (z > 1e-5f && un >= 0.f && un <= 1.f && vn >= 0.f && vn <= 1.f)
                   ? 1.f : 0.f;
    ref_cam[(size_t)idx * 2 + 0] = un;
    ref_cam[(size_t)idx * 2 + 1] = vn;
    valid[idx] = ok;
}

// ---------------------------------------------------------------------------
// TSA deformable sampling (prev & cur fused, averaged). Thread = (query, head).
// Output written in f16 (consumed only by the projection GEMM).
// ---------------------------------------------------------------------------
__global__ __launch_bounds__(128)
void tsa_deform(const float* __restrict__ vprev, const float* __restrict__ vcur,
                const float* __restrict__ off,  const float* __restrict__ attn,
                _Float16* __restrict__ outp)
{
    int idx = blockIdx.x * 128 + threadIdx.x;
    if (idx >= NQ * HEADS) return;
    int h = idx & (HEADS - 1);
    int n = idx >> 3;
    int qx = n & (BEVW - 1), qy = n >> 7;
    float rx = (qx + 0.5f) / (float)BEVW;
    float ry = (qy + 0.5f) / (float)BEVH;

    const float* ap = attn + (size_t)n * HP + h * NPTS;
    float mx = fmaxf(fmaxf(ap[0], ap[1]), fmaxf(ap[2], ap[3]));
    float w[NPTS], sum = 0.f;
    #pragma unroll
    for (int p = 0; p < NPTS; p++) { w[p] = expf(ap[p] - mx); sum += w[p]; }
    float rs = 1.f / sum;

    float out[HD];
    #pragma unroll
    for (int d = 0; d < HD; d++) out[d] = 0.f;

    const float* op = off + (size_t)n * HP2 + h * NPTS * 2;
    #pragma unroll
    for (int p = 0; p < NPTS; p++) {
        float wp = w[p] * rs;
        float lx = rx + op[p * 2 + 0] * (1.f / BEVW);
        float ly = ry + op[p * 2 + 1] * (1.f / BEVH);
        float ix = lx * (float)BEVW - 0.5f;
        float iy = ly * (float)BEVH - 0.5f;
        float x0 = floorf(ix), y0 = floorf(iy);
        float fx = ix - x0, fy = iy - y0;
        for (int dy = 0; dy < 2; dy++) {
            for (int dx = 0; dx < 2; dx++) {
                float xi = x0 + dx, yi = y0 + dy;
                if (xi < 0.f || xi > (float)(BEVW - 1) ||
                    yi < 0.f || yi > (float)(BEVH - 1)) continue;
                float wc = (dx ? fx : 1.f - fx) * (dy ? fy : 1.f - fy) * wp;
                size_t cell = ((size_t)((int)yi * BEVW + (int)xi)) * CDIM + h * HD;
                const float4* sp = (const float4*)(vprev + cell);
                const float4* sc = (const float4*)(vcur + cell);
                #pragma unroll
                for (int d4 = 0; d4 < HD / 4; d4++) {
                    float4 a = sp[d4], b = sc[d4];
                    out[d4*4+0] += wc * (a.x + b.x);
                    out[d4*4+1] += wc * (a.y + b.y);
                    out[d4*4+2] += wc * (a.z + b.z);
                    out[d4*4+3] += wc * (a.w + b.w);
                }
            }
        }
    }
    _Float16* dst = outp + (size_t)n * CDIM + h * HD;
    #pragma unroll
    for (int d = 0; d < HD; d++) dst[d] = (_Float16)(0.5f * out[d]);
}

// ---------------------------------------------------------------------------
// Spatial cross-attention sampling. Thread = (query, head). f16 output.
// ---------------------------------------------------------------------------
__global__ __launch_bounds__(128)
void sca_sample(const float* __restrict__ vimg, const float* __restrict__ off,
                const float* __restrict__ attn, const float* __restrict__ refc,
                const float* __restrict__ valid, _Float16* __restrict__ outp)
{
    int idx = blockIdx.x * 128 + threadIdx.x;
    if (idx >= NQ * HEADS) return;
    int h = idx & (HEADS - 1);
    int n = idx >> 3;

    const float* ap = attn + (size_t)n * HP + h * NPTS;
    float mx = fmaxf(fmaxf(ap[0], ap[1]), fmaxf(ap[2], ap[3]));
    float w[NPTS], sum = 0.f;
    #pragma unroll
    for (int p = 0; p < NPTS; p++) { w[p] = expf(ap[p] - mx); sum += w[p]; }
    float rs = 1.f / sum;

    const float* op = off + (size_t)n * HP2 + h * NPTS * 2;
    float out[HD];
    #pragma unroll
    for (int d = 0; d < HD; d++) out[d] = 0.f;
    float cnt = 0.f;

    for (int c = 0; c < NCAM; c++) {
        const float* vb = vimg + (size_t)c * HF * WF * CDIM;
        #pragma unroll
        for (int p = 0; p < NPTS; p++) {
            size_t vi = ((size_t)c * NQ + n) * NPTS + p;
            float vl = valid[vi];
            cnt += vl;
            if (vl == 0.f) continue;
            float lx = refc[vi * 2 + 0] + op[p * 2 + 0] * (1.f / WF);
            float ly = refc[vi * 2 + 1] + op[p * 2 + 1] * (1.f / HF);
            float wgt = w[p] * rs * vl;
            float ix = lx * (float)WF - 0.5f;
            float iy = ly * (float)HF - 0.5f;
            float x0 = floorf(ix), y0 = floorf(iy);
            float fx = ix - x0, fy = iy - y0;
            for (int dy = 0; dy < 2; dy++) {
                for (int dx = 0; dx < 2; dx++) {
                    float xi = x0 + dx, yi = y0 + dy;
                    if (xi < 0.f || xi > (float)(WF - 1) ||
                        yi < 0.f || yi > (float)(HF - 1)) continue;
                    float wc = (dx ? fx : 1.f - fx) * (dy ? fy : 1.f - fy) * wgt;
                    size_t cell = ((size_t)((int)yi * WF + (int)xi)) * CDIM + h * HD;
                    const float4* sp = (const float4*)(vb + cell);
                    #pragma unroll
                    for (int d4 = 0; d4 < HD / 4; d4++) {
                        float4 a = sp[d4];
                        out[d4*4+0] += wc * a.x;
                        out[d4*4+1] += wc * a.y;
                        out[d4*4+2] += wc * a.z;
                        out[d4*4+3] += wc * a.w;
                    }
                }
            }
        }
    }
    float rc = 1.f / fmaxf(cnt, 1.f);
    _Float16* dst = outp + (size_t)n * CDIM + h * HD;
    #pragma unroll
    for (int d = 0; d < HD; d++) dst[d] = (_Float16)(out[d] * rc);
}

// ---------------------------------------------------------------------------
// q = LayerNorm(q + t) * g + b ; writes f32 q and an f16 shadow copy for GEMMs.
// One wave32 per 256-wide row.
// ---------------------------------------------------------------------------
__global__ __launch_bounds__(256)
void add_layernorm(float* __restrict__ q, _Float16* __restrict__ qh,
                   const float* __restrict__ t,
                   const float* __restrict__ g, const float* __restrict__ b)
{
    int row  = blockIdx.x * 8 + (threadIdx.x >> 5);
    int lane = threadIdx.x & 31;
    const float* qr = q + (size_t)row * CDIM;
    const float* tr = t + (size_t)row * CDIM;
    float x[8], s = 0.f, ss = 0.f;
    #pragma unroll
    for (int i = 0; i < 8; i++) {
        int c = lane + i * 32;
        float v = qr[c] + tr[c];
        x[i] = v; s += v; ss += v * v;
    }
    #pragma unroll
    for (int d = 16; d > 0; d >>= 1) {
        s  += __shfl_xor(s,  d, 32);
        ss += __shfl_xor(ss, d, 32);
    }
    float mean = s * (1.f / CDIM);
    float var  = ss * (1.f / CDIM) - mean * mean;
    float inv  = rsqrtf(var + 1e-5f);
    #pragma unroll
    for (int i = 0; i < 8; i++) {
        int c = lane + i * 32;
        float v = (x[i] - mean) * inv * g[c] + b[c];
        q[(size_t)row * CDIM + c]  = v;
        qh[(size_t)row * CDIM + c] = (_Float16)v;
    }
}

// ---------------------------------------------------------------------------
// q (Nq, C) -> out (1, C, H, W)
// ---------------------------------------------------------------------------
__global__ __launch_bounds__(256)
void write_out(const float* __restrict__ q, float* __restrict__ out)
{
    int idx = blockIdx.x * 256 + threadIdx.x;
    if (idx >= NQ * CDIM) return;
    int n  = idx & (NQ - 1);
    int ch = idx >> 14;
    out[idx] = q[(size_t)n * CDIM + ch];
}

// ---------------------------------------------------------------------------
// transposed-f16 weight arena offsets (per layer, element counts)
// ---------------------------------------------------------------------------
enum : size_t {
    OFF_VW  = 0,
    OFF_OW  = OFF_VW  + 256 * 256,
    OFF_AW  = OFF_OW  + 256 * 64,
    OFF_PW  = OFF_AW  + 256 * 32,
    OFF_SVW = OFF_PW  + 256 * 256,
    OFF_SOW = OFF_SVW + 256 * 256,
    OFF_SAW = OFF_SOW + 256 * 64,
    OFF_SPW = OFF_SAW + 256 * 32,
    OFF_W1  = OFF_SPW + 256 * 256,
    OFF_W2  = OFF_W1  + 256 * 512,
    WLAYER  = OFF_W2  + 512 * 256,
};

// ---------------------------------------------------------------------------
extern "C" void kernel_launch(void* const* d_in, const int* in_sizes, int n_in,
                              void* d_out, int out_size, void* d_ws, size_t ws_size,
                              hipStream_t stream)
{
    (void)in_sizes; (void)n_in; (void)out_size; (void)ws_size;

    const float* image_feats = (const float*)d_in[0];
    const float* cam_intr    = (const float*)d_in[1];
    const float* cam_extr    = (const float*)d_in[2];
    const float* prev_bev    = (const float*)d_in[3];
    const float* bev_embed   = (const float*)d_in[4];
    const float* tsa_vw = (const float*)d_in[5];
    const float* tsa_vb = (const float*)d_in[6];
    const float* tsa_ow = (const float*)d_in[7];
    const float* tsa_ob = (const float*)d_in[8];
    const float* tsa_aw = (const float*)d_in[9];
    const float* tsa_ab = (const float*)d_in[10];
    const float* tsa_pw = (const float*)d_in[11];
    const float* tsa_pb = (const float*)d_in[12];
    const float* sca_vw = (const float*)d_in[13];
    const float* sca_vb = (const float*)d_in[14];
    const float* sca_ow = (const float*)d_in[15];
    const float* sca_ob = (const float*)d_in[16];
    const float* sca_aw = (const float*)d_in[17];
    const float* sca_ab = (const float*)d_in[18];
    const float* sca_pw = (const float*)d_in[19];
    const float* sca_pb = (const float*)d_in[20];
    const float* ffn_w1 = (const float*)d_in[21];
    const float* ffn_b1 = (const float*)d_in[22];
    const float* ffn_w2 = (const float*)d_in[23];
    const float* ffn_b2 = (const float*)d_in[24];
    const float* ln1_g  = (const float*)d_in[25];
    const float* ln1_b  = (const float*)d_in[26];
    const float* ln2_g  = (const float*)d_in[27];
    const float* ln2_b  = (const float*)d_in[28];
    const float* ln3_g  = (const float*)d_in[29];
    const float* ln3_b  = (const float*)d_in[30];

    // -------- workspace layout (byte allocator, 256B aligned blocks) --------
    char* wsb = (char*)d_ws;
    size_t off = 0;
    auto alloc = [&](size_t bytes) -> void* {
        void* p = wsb + off;
        off += (bytes + 255) & ~(size_t)255;
        return p;
    };
    float*    q_buf     = (float*)   alloc((size_t)NQ * CDIM * 4);
    float*    v_prev    = (float*)   alloc((size_t)NQ * CDIM * 4);
    float*    v_cur     = (float*)   alloc((size_t)NQ * CDIM * 4);
    float*    t_buf     = (float*)   alloc((size_t)NQ * CDIM * 4);
    float*    off_buf   = (float*)   alloc((size_t)NQ * HP2 * 4);
    float*    attn_buf  = (float*)   alloc((size_t)NQ * HP * 4);
    float*    v_img     = (float*)   alloc((size_t)FEAT_PAD * CDIM * 4);
    float*    ref_cam   = (float*)   alloc((size_t)NCAM * NQ * NPTS * 2 * 4);
    float*    valid_buf = (float*)   alloc((size_t)NCAM * NQ * NPTS * 4);
    float*    ego2cam   = (float*)   alloc((size_t)NCAM * 16 * 4);
    _Float16* prev_h    = (_Float16*)alloc((size_t)NQ * CDIM * 2);
    _Float16* q_h       = (_Float16*)alloc((size_t)NQ * CDIM * 2);
    _Float16* a_h       = (_Float16*)alloc((size_t)NQ * CDIM * 2);
    _Float16* feats_h   = (_Float16*)alloc((size_t)FEAT_PAD * CDIM * 2);
    _Float16* wts       = (_Float16*)alloc((size_t)WLAYER * NLAYERS * 2);
    _Float16* ffn_mid_h = (_Float16*)v_img;   // reuse (dead after sca_sample)

    const int C = CDIM, C2 = 2 * CDIM;

    // -------- one-time setup --------
    (void)hipMemcpyAsync(q_buf, bev_embed, (size_t)NQ * C * sizeof(float),
                         hipMemcpyDeviceToDevice, stream);
    cast_f32_to_f16<<<(NQ * C + 255) / 256, 256, 0, stream>>>(bev_embed, q_h, NQ * C);
    cast_f32_to_f16<<<(NQ * C + 255) / 256, 256, 0, stream>>>(prev_bev, prev_h, NQ * C);
    feats_to_flat_h<<<(FEAT_PAD * CDIM + 255) / 256, 256, 0, stream>>>(image_feats, feats_h);
    invert_ext<<<1, NCAM, 0, stream>>>(cam_extr, ego2cam);
    project_refs<<<(NCAM * NQ * NPTS + 255) / 256, 256, 0, stream>>>(
        ego2cam, cam_intr, ref_cam, valid_buf);

    // pre-transpose + cast all weights to f16 [N][K]
    auto cast_w = [&](const float* W, _Float16* Wt, int K, int N) {
        int tot = K * N;
        cast_transpose_w<<<(tot + 255) / 256, 256, 0, stream>>>(W, Wt, K, N);
    };
    for (int l = 0; l < NLAYERS; l++) {
        _Float16* wl = wts + (size_t)l * WLAYER;
        cast_w(tsa_vw + (size_t)l*C*C,   wl + OFF_VW,  C,  C);
        cast_w(tsa_ow + (size_t)l*C*HP2, wl + OFF_OW,  C,  HP2);
        cast_w(tsa_aw + (size_t)l*C*HP,  wl + OFF_AW,  C,  HP);
        cast_w(tsa_pw + (size_t)l*C*C,   wl + OFF_PW,  C,  C);
        cast_w(sca_vw + (size_t)l*C*C,   wl + OFF_SVW, C,  C);
        cast_w(sca_ow + (size_t)l*C*HP2, wl + OFF_SOW, C,  HP2);
        cast_w(sca_aw + (size_t)l*C*HP,  wl + OFF_SAW, C,  HP);
        cast_w(sca_pw + (size_t)l*C*C,   wl + OFF_SPW, C,  C);
        cast_w(ffn_w1 + (size_t)l*C*C2,  wl + OFF_W1,  C,  C2);
        cast_w(ffn_w2 + (size_t)l*C2*C,  wl + OFF_W2,  C2, C);
    }

    const int samp_blocks = (NQ * HEADS + 127) / 128;

    for (int l = 0; l < NLAYERS; l++) {
        _Float16* wl = wts + (size_t)l * WLAYER;

        // ---- temporal self-attention ----
        run_gemm(prev_h, wl + OFF_VW, tsa_vb + l*C,   v_prev,   nullptr, NQ, C,   C, 0, stream);
        run_gemm(q_h,    wl + OFF_VW, tsa_vb + l*C,   v_cur,    nullptr, NQ, C,   C, 0, stream);
        run_gemm(q_h,    wl + OFF_OW, tsa_ob + l*HP2, off_buf,  nullptr, NQ, HP2, C, 0, stream);
        run_gemm(q_h,    wl + OFF_AW, tsa_ab + l*HP,  attn_buf, nullptr, NQ, HP,  C, 0, stream);
        tsa_deform<<<samp_blocks, 128, 0, stream>>>(v_prev, v_cur, off_buf, attn_buf, a_h);
        run_gemm(a_h, wl + OFF_PW, tsa_pb + l*C, t_buf, nullptr, NQ, C, C, 0, stream);
        add_layernorm<<<NQ / 8, 256, 0, stream>>>(q_buf, q_h, t_buf, ln1_g + l*C, ln1_b + l*C);

        // ---- spatial cross-attention ----
        run_gemm(feats_h, wl + OFF_SVW, sca_vb + l*C,   v_img,    nullptr, FEAT_PAD, C,   C, 0, stream);
        run_gemm(q_h,     wl + OFF_SOW, sca_ob + l*HP2, off_buf,  nullptr, NQ,       HP2, C, 0, stream);
        run_gemm(q_h,     wl + OFF_SAW, sca_ab + l*HP,  attn_buf, nullptr, NQ,       HP,  C, 0, stream);
        sca_sample<<<samp_blocks, 128, 0, stream>>>(v_img, off_buf, attn_buf,
                                                    ref_cam, valid_buf, a_h);
        run_gemm(a_h, wl + OFF_SPW, sca_pb + l*C, t_buf, nullptr, NQ, C, C, 0, stream);
        add_layernorm<<<NQ / 8, 256, 0, stream>>>(q_buf, q_h, t_buf, ln2_g + l*C, ln2_b + l*C);

        // ---- FFN ----
        run_gemm(q_h,       wl + OFF_W1, ffn_b1 + l*C2, nullptr, ffn_mid_h, NQ, C2, C,  1, stream);
        run_gemm(ffn_mid_h, wl + OFF_W2, ffn_b2 + l*C,  t_buf,   nullptr,   NQ, C,  C2, 0, stream);
        add_layernorm<<<NQ / 8, 256, 0, stream>>>(q_buf, q_h, t_buf, ln3_g + l*C, ln3_b + l*C);
    }

    write_out<<<(NQ * CDIM + 255) / 256, 256, 0, stream>>>(q_buf, (float*)d_out);
}
